// InteractionLSTM_Cell_53128745452149
// MI455X (gfx1250) — compile-verified, matrix operand order checked by reading, not run on hardware
//
#include <hip/hip_runtime.h>
#include <hip/hip_bf16.h>
#include <math.h>

// ---------------- problem dims ----------------
#define TT    15
#define NN    8
#define DD    64
#define UU    2056
#define HH    512
#define BUFD  1680
#define XDIM  2704          // BUFD + 2*HH
#define GIN   4760          // XDIM + UU
#define MROWS 2048          // B*N
#define PROWS 16384         // B*N*N
#define NG    8224          // 4*UU

typedef __attribute__((ext_vector_type(16))) __bf16 bf16x16;
typedef __attribute__((ext_vector_type(8)))  float  f32x8;

struct alignas(16) U4 { unsigned int x, y, z, w; };
union Frag16 { bf16x16 v; __bf16 e[16]; U4 q[2]; };
union H8     { __bf16 h[8]; U4 q; };

// ---------------- device helpers ----------------
__device__ __forceinline__ float gelu_exact(float x) {
    return 0.5f * x * (1.0f + erff(x * 0.70710678118654752440f));
}
__device__ __forceinline__ float sigmoid_f(float x) {
    return 1.0f / (1.0f + expf(-x));
}

// =====================================================================
// bf16 WMMA GEMM:  Out(M,N) = act(A(M,K) @ B(K,N) + bias) [+ addSrc]
//  - A: bf16 row-major (lda), staged in LDS, ping-pong double-buffered
//  - B: bf16 PRE-SWIZZLED into WMMA fragment tiles [nt][kt][lane][16]
//  - block tile 128x64, 4 waves, each wave 32x64 (2 M-strips x 4 N-tiles)
//    => 8 WMMAs per wave per 32-K step
// =====================================================================
#define GEMM_BM 128
#define GEMM_BN 64
#define GEMM_BK 32
#define GEMM_THREADS 128
#define A_PITCH 40           // halfwords; 80B row pitch keeps 8-hw runs 16B aligned

__device__ __forceinline__ void loadA_tile(U4 pf[4], const __bf16* __restrict__ A, int lda,
                                           int M, int K, int m0, int k0, int tid)
{
    if (m0 + GEMM_BM <= M && k0 + GEMM_BK <= K) {
#pragma unroll
        for (int s = 0; s < 4; ++s) {
            int idx = tid + s * GEMM_THREADS;
            int r = idx >> 2, c = (idx & 3) << 3;        // 4 x b128 per 32-elem row
            pf[s] = *(const U4*)(A + (size_t)(m0 + r) * lda + k0 + c);
        }
    } else {
#pragma unroll
        for (int s = 0; s < 4; ++s) {
            int idx = tid + s * GEMM_THREADS;
            int r = idx >> 2, c = (idx & 3) << 3;
            H8 t;
#pragma unroll
            for (int q = 0; q < 8; ++q) {
                int gm = m0 + r, gk = k0 + c + q;
                t.h[q] = (gm < M && gk < K) ? A[(size_t)gm * lda + gk] : (__bf16)0.0f;
            }
            pf[s] = t.q;
        }
    }
}

template<bool OUT_BF16>
__global__ __launch_bounds__(GEMM_THREADS)
void wmma_gemm2_k(const __bf16* __restrict__ A, int lda,
                  const __bf16* __restrict__ Bsw, int KT,
                  void* __restrict__ Out, int ldo,
                  const float* __restrict__ bias,
                  const float* __restrict__ addSrc, int ldadd,
                  int M, int N, int K, int act)
{
    __shared__ __align__(16) __bf16 As[2][GEMM_BM][A_PITCH];

    const int tid     = threadIdx.x;
    const int wave    = tid >> 5;
    const int lane    = tid & 31;
    const int lane_lo = lane & 15;
    const int kg      = lane >> 4;
    const int m0      = blockIdx.y * GEMM_BM;
    const int ntb     = blockIdx.x * 4;          // first 16-col tile of this block

    f32x8 acc[2][4];
#pragma unroll
    for (int st = 0; st < 2; ++st)
#pragma unroll
        for (int t = 0; t < 4; ++t)
#pragma unroll
            for (int i = 0; i < 8; ++i) acc[st][t][i] = 0.0f;

    // prologue: stage K-tile 0
    {
        U4 pf[4];
        loadA_tile(pf, A, lda, M, K, m0, 0, tid);
#pragma unroll
        for (int s = 0; s < 4; ++s) {
            int idx = tid + s * GEMM_THREADS;
            int r = idx >> 2, c = (idx & 3) << 3;
            *(U4*)&As[0][r][c] = pf[s];
        }
    }
    __syncthreads();

    for (int step = 0; step < KT; ++step) {
        const int  cur    = step & 1;
        const bool havePf = (step + 1 < KT);
        U4 pf[4];
        if (havePf) loadA_tile(pf, A, lda, M, K, m0, (step + 1) * GEMM_BK, tid);

        // A fragments: two 16B ds loads each (runs are 16B aligned)
        Frag16 af0, af1;
        {
            const __bf16* a0 = &As[cur][wave * 32 + lane_lo][kg * 8];
            af0.q[0] = *(const U4*)(a0);
            af0.q[1] = *(const U4*)(a0 + 16);
            const __bf16* a1 = &As[cur][wave * 32 + 16 + lane_lo][kg * 8];
            af1.q[0] = *(const U4*)(a1);
            af1.q[1] = *(const U4*)(a1 + 16);
        }
#pragma unroll
        for (int t = 0; t < 4; ++t) {
            Frag16 bf;
            const __bf16* bp = Bsw + (((size_t)(ntb + t) * KT + step) << 9) + (lane << 4);
            bf.q[0] = *(const U4*)(bp);
            bf.q[1] = *(const U4*)(bp + 8);
            acc[0][t] = __builtin_amdgcn_wmma_f32_16x16x32_bf16(
                false, af0.v, false, bf.v, (short)0, acc[0][t], false, false);
            acc[1][t] = __builtin_amdgcn_wmma_f32_16x16x32_bf16(
                false, af1.v, false, bf.v, (short)0, acc[1][t], false, false);
        }
        if (havePf) {
#pragma unroll
            for (int s = 0; s < 4; ++s) {
                int idx = tid + s * GEMM_THREADS;
                int r = idx >> 2, c = (idx & 3) << 3;
                *(U4*)&As[cur ^ 1][r][c] = pf[s];
            }
        }
        __syncthreads();
    }

    // epilogue: C layout VGPR i -> M = i + 8*kg, N = lane_lo
#pragma unroll
    for (int st = 0; st < 2; ++st)
#pragma unroll
        for (int t = 0; t < 4; ++t)
#pragma unroll
            for (int i = 0; i < 8; ++i) {
                int gm = m0 + wave * 32 + st * 16 + kg * 8 + i;
                int gn = blockIdx.x * GEMM_BN + t * 16 + lane_lo;
                if (gm < M && gn < N) {
                    float v = acc[st][t][i];
                    if (bias) v += bias[gn];
                    if (act == 1)      v = (v > 0.0f) ? v : expm1f(v);   // elu
                    else if (act == 2) v = gelu_exact(v);                // erf-gelu
                    if (addSrc) v += addSrc[(size_t)gm * ldadd + gn];
                    if (OUT_BF16) ((__bf16*)Out)[(size_t)gm * ldo + gn] = (__bf16)v;
                    else          ((float*) Out)[(size_t)gm * ldo + gn] = v;
                }
            }
}

// ---------------- B-matrix swizzle (fp32 -> bf16 fragment tiles) ----------------
// Destination layout: [nt][kt][lane 0..31][halfword 0..15], zero-padded.
// Logical B may be a concatenation of up to 4 column segments.
struct BSrc { const float* p[4]; int width[4]; int sld[4]; int nseg; };

__global__ void cvt_swz_k(__bf16* __restrict__ dst, BSrc s,
                          int K, int N, int KT, long long total)
{
    long long e = (long long)blockIdx.x * blockDim.x + threadIdx.x;
    if (e >= total) return;
    int h        = (int)(e & 15);
    int lane     = (int)((e >> 4) & 31);
    long long tl = e >> 9;
    int kt = (int)(tl % KT);
    int nt = (int)(tl / KT);
    int n  = nt * 16 + (lane & 15);
    int kgrp = lane >> 4;
    int k  = kt * 32 + ((h & 8) ? 16 : 0) + kgrp * 8 + (h & 7);
    float v = 0.0f;
    if (k < K && n < N) {
        int base = 0;
#pragma unroll
        for (int j = 0; j < 4; ++j) {
            if (j < s.nseg) {
                int w = s.width[j];
                if (n < base + w) { v = s.p[j][(size_t)k * s.sld[j] + (n - base)]; break; }
                base += w;
            }
        }
    }
    dst[e] = (__bf16)v;
}

// ---------------- elementwise / pack kernels ----------------
__global__ void cvt_strided_k(__bf16* __restrict__ dst, long long dld,
                              const float* __restrict__ src, long long sld,
                              long long R, long long C)
{
    long long i = (long long)blockIdx.x * blockDim.x + threadIdx.x;
    if (i >= R * C) return;
    long long r = i / C, c = i % C;
    dst[r * dld + c] = (__bf16)src[r * sld + c];
}

__global__ void gather_zb_k(__bf16* __restrict__ dst, const float* __restrict__ z)
{
    long long i = (long long)blockIdx.x * blockDim.x + threadIdx.x;
    if (i >= (long long)MROWS * (DD * TT)) return;
    int m = (int)(i / (DD * TT));
    int k = (int)(i % (DD * TT));
    int b = m >> 3, n = m & 7;
    int d = k / TT, t = k % TT;
    dst[i] = (__bf16)z[(((size_t)b * TT + t) * NN + n) * DD + d];
}

__global__ void pairwise_k(__bf16* __restrict__ r1, __bf16* __restrict__ l1,
                           const float* __restrict__ AAL,   // (2048,1536) = ai|aj|al
                           const float* __restrict__ bl,    // (2048,512)
                           const float* __restrict__ b_r1, const float* __restrict__ b_l1)
{
    long long idx = (long long)blockIdx.x * blockDim.x + threadIdx.x;
    if (idx >= (long long)PROWS * HH) return;
    int row = (int)(idx >> 9);
    int h   = (int)(idx & (HH - 1));
    int j   = row & 7;
    int bi  = row >> 3;
    int b   = bi >> 3;
    float rv = AAL[(size_t)bi * 1536 + h] + AAL[(size_t)(b * 8 + j) * 1536 + 512 + h] + b_r1[h];
    float lv = AAL[(size_t)bi * 1536 + 1024 + h] + bl[(size_t)(b * 8 + j) * HH + h] + b_l1[h];
    r1[idx] = (__bf16)gelu_exact(rv);
    l1[idx] = (__bf16)gelu_exact(lv);
}

__global__ void reduce_inter_k(__bf16* __restrict__ xh, const float* __restrict__ inter)
{
    long long idx = (long long)blockIdx.x * blockDim.x + threadIdx.x;
    if (idx >= (long long)MROWS * HH) return;
    int bi = (int)(idx >> 9);
    int h  = (int)(idx & (HH - 1));
    float s = 0.0f, mx = -3.4e38f;
#pragma unroll
    for (int j = 0; j < 8; ++j) {
        float v = inter[((size_t)bi * 8 + j) * HH + h];
        s += v; mx = fmaxf(mx, v);
    }
    xh[(size_t)bi * GIN + BUFD + h]      = (__bf16)s;
    xh[(size_t)bi * GIN + BUFD + HH + h] = (__bf16)mx;
}

__global__ void lstm_point_k(float* __restrict__ Cout, float* __restrict__ Hout,
                             __bf16* __restrict__ Hbf,
                             const float* __restrict__ G,      // (2048,8224) = I|F|O|C
                             const float* __restrict__ cells,
                             const float* __restrict__ bI, const float* __restrict__ bF,
                             const float* __restrict__ bO, const float* __restrict__ bC)
{
    long long idx = (long long)blockIdx.x * blockDim.x + threadIdx.x;
    if (idx >= (long long)MROWS * UU) return;
    int m = (int)(idx / UU);
    int u = (int)(idx % UU);
    const float* g = G + (size_t)m * NG;
    float gI = sigmoid_f(g[u]           + bI[u]);
    float gF = sigmoid_f(g[UU + u]      + bF[u]);
    float gO = sigmoid_f(g[2 * UU + u]  + bO[u]);
    float ct = tanhf(    g[3 * UU + u]  + bC[u]);
    float c  = gF * cells[idx] + gI * ct;
    float hn = gO * tanhf(c);
    Cout[idx] = c;
    Hout[idx] = hn;
    Hbf[idx]  = (__bf16)hn;
}

// ---------------- host-side helpers ----------------
static inline int swz_KT(int K)  { return (K + 31) / 32; }
static inline int swz_NTp(int N) { return ((N + 63) / 64) * 4; }
static inline size_t swz_bytes(int K, int N) {
    return (size_t)swz_NTp(N) * swz_KT(K) * 512 * 2;
}

static void launch_swz(__bf16* dst, int K, int N, const BSrc& s, hipStream_t str)
{
    int KT = swz_KT(K);
    long long total = (long long)swz_NTp(N) * KT * 512;
    cvt_swz_k<<<(int)((total + 255) / 256), 256, 0, str>>>(dst, s, K, N, KT, total);
}

static BSrc seg1(const float* p, int w, int ld)
{
    BSrc s{}; s.p[0] = p; s.width[0] = w; s.sld[0] = ld; s.nseg = 1; return s;
}

static void launch_gemm(bool outBf16,
                        const __bf16* A, int lda, const __bf16* Bsw,
                        void* out, int ldo, const float* bias,
                        const float* addSrc, int ldadd,
                        int M, int N, int K, int act, hipStream_t s)
{
    dim3 g((N + GEMM_BN - 1) / GEMM_BN, (M + GEMM_BM - 1) / GEMM_BM);
    int KT = swz_KT(K);
    if (outBf16)
        wmma_gemm2_k<true><<<g, GEMM_THREADS, 0, s>>>(A, lda, Bsw, KT, out, ldo, bias, addSrc, ldadd, M, N, K, act);
    else
        wmma_gemm2_k<false><<<g, GEMM_THREADS, 0, s>>>(A, lda, Bsw, KT, out, ldo, bias, addSrc, ldadd, M, N, K, act);
}

static void launch_cvt(__bf16* dst, long long dld, const float* src, long long sld,
                       long long R, long long C, hipStream_t s)
{
    long long total = R * C;
    cvt_strided_k<<<(int)((total + 255) / 256), 256, 0, s>>>(dst, dld, src, sld, R, C);
}

extern "C" void kernel_launch(void* const* d_in, const int* in_sizes, int n_in,
                              void* d_out, int out_size, void* d_ws, size_t ws_size,
                              hipStream_t stream)
{
    (void)in_sizes; (void)n_in; (void)out_size;

    const float* z      = (const float*)d_in[0];
    const float* Cells  = (const float*)d_in[1];
    const float* hidden = (const float*)d_in[2];
    const float* W_buf  = (const float*)d_in[3];
    const float* b_buf  = (const float*)d_in[4];
    const float* W_r1   = (const float*)d_in[5];
    const float* b_r1   = (const float*)d_in[6];
    const float* W_r2   = (const float*)d_in[7];
    const float* b_r2   = (const float*)d_in[8];
    const float* W_r3   = (const float*)d_in[9];
    const float* b_r3   = (const float*)d_in[10];
    const float* W_l1   = (const float*)d_in[11];
    const float* b_l1   = (const float*)d_in[12];
    const float* W_l2   = (const float*)d_in[13];
    const float* b_l2   = (const float*)d_in[14];
    const float* W_l3   = (const float*)d_in[15];
    const float* b_l3   = (const float*)d_in[16];
    const float* W_I    = (const float*)d_in[17];
    const float* b_I    = (const float*)d_in[18];
    const float* W_F    = (const float*)d_in[19];
    const float* b_F    = (const float*)d_in[20];
    const float* W_O    = (const float*)d_in[21];
    const float* b_O    = (const float*)d_in[22];
    const float* W_C    = (const float*)d_in[23];
    const float* b_C    = (const float*)d_in[24];
    const float* W_d    = (const float*)d_in[25];
    const float* b_d    = (const float*)d_in[26];

    float* out_dec = (float*)d_out;
    float* out_C   = out_dec + (size_t)MROWS * DD;
    float* out_H   = out_C   + (size_t)MROWS * UU;

    // ---- workspace carve ----
    char* ws = (char*)d_ws;
    size_t off = 0;
    auto carve = [&](size_t bytes) -> char* {
        size_t o = (off + 255) & ~(size_t)255;
        off = o + bytes;
        return ws + o;
    };
    __bf16* zb_bf    = (__bf16*)carve((size_t)MROWS * DD * TT * 2);   // 2048x960 (A)
    __bf16* xh_bf    = (__bf16*)carve((size_t)MROWS * GIN * 2);       // 2048x4760 (A)
    __bf16* cells_bf = (__bf16*)carve((size_t)MROWS * UU * 2);        // 2048x2056 (A)
    float*  AAL      = (float*) carve((size_t)MROWS * 1536 * 4);      // ai|aj|al
    float*  bl_f     = (float*) carve((size_t)MROWS * HH * 4);
    __bf16* r1_bf    = (__bf16*)carve((size_t)PROWS * HH * 2);
    __bf16* l1_bf    = (__bf16*)carve((size_t)PROWS * HH * 2);
    __bf16* r2_bf    = (__bf16*)carve((size_t)PROWS * HH * 2);
    __bf16* l2_bf    = (__bf16*)carve((size_t)PROWS * HH * 2);
    float*  inter_f  = (float*) carve((size_t)PROWS * HH * 4);
    float*  G_f      = (float*) carve((size_t)MROWS * NG * 4);
    __bf16* Hn_bf    = (__bf16*)carve((size_t)MROWS * UU * 2);
    // swizzled B (weights)
    __bf16* Wbuf_sw  = (__bf16*)carve(swz_bytes(DD * TT, BUFD));
    __bf16* Wpk1_sw  = (__bf16*)carve(swz_bytes(UU, 1536));
    __bf16* Wl1b_sw  = (__bf16*)carve(swz_bytes(BUFD, HH));
    __bf16* Wr2_sw   = (__bf16*)carve(swz_bytes(HH, HH));
    __bf16* Wr3_sw   = (__bf16*)carve(swz_bytes(HH, HH));
    __bf16* Wl2_sw   = (__bf16*)carve(swz_bytes(HH, HH));
    __bf16* Wl3_sw   = (__bf16*)carve(swz_bytes(HH, HH));
    __bf16* Wg_sw    = (__bf16*)carve(swz_bytes(GIN, NG));
    __bf16* Wd_sw    = (__bf16*)carve(swz_bytes(UU, DD));
    if (off > ws_size) return;   // deterministic no-op if workspace too small

    // ---- stage 0: activations -> bf16 ----
    {
        long long tot = (long long)MROWS * DD * TT;
        gather_zb_k<<<(int)((tot + 255) / 256), 256, 0, stream>>>(zb_bf, z);
    }
    launch_cvt(cells_bf, UU, Cells, UU, MROWS, UU, stream);
    launch_cvt(xh_bf + XDIM, GIN, hidden, UU, MROWS, UU, stream);   // hidden tail of xh

    // ---- stage 0b: weights -> swizzled bf16 fragment tiles ----
    launch_swz(Wbuf_sw, DD * TT, BUFD, seg1(W_buf, BUFD, BUFD), stream);
    {   // [W_r1[:U] | W_r1[U:] | W_l1[:U]]  (K=2056, N=1536)
        BSrc s{};
        s.p[0] = W_r1;                     s.width[0] = HH; s.sld[0] = HH;
        s.p[1] = W_r1 + (size_t)UU * HH;   s.width[1] = HH; s.sld[1] = HH;
        s.p[2] = W_l1;                     s.width[2] = HH; s.sld[2] = HH;
        s.nseg = 3;
        launch_swz(Wpk1_sw, UU, 1536, s, stream);
    }
    launch_swz(Wl1b_sw, BUFD, HH, seg1(W_l1 + (size_t)UU * HH, HH, HH), stream);
    launch_swz(Wr2_sw, HH, HH, seg1(W_r2, HH, HH), stream);
    launch_swz(Wr3_sw, HH, HH, seg1(W_r3, HH, HH), stream);
    launch_swz(Wl2_sw, HH, HH, seg1(W_l2, HH, HH), stream);
    launch_swz(Wl3_sw, HH, HH, seg1(W_l3, HH, HH), stream);
    {   // [W_I | W_F | W_O | W_C]  (K=4760, N=8224)
        BSrc s{};
        s.p[0] = W_I; s.p[1] = W_F; s.p[2] = W_O; s.p[3] = W_C;
        for (int j = 0; j < 4; ++j) { s.width[j] = UU; s.sld[j] = UU; }
        s.nseg = 4;
        launch_swz(Wg_sw, GIN, NG, s, stream);
    }
    launch_swz(Wd_sw, UU, DD, seg1(W_d, DD, DD), stream);

    // ---- G1: z_b = elu(zb @ W_buf + b_buf) -> xh[:,0:1680] (bf16) ----
    launch_gemm(true, zb_bf, DD * TT, Wbuf_sw, xh_bf, GIN,
                b_buf, nullptr, 0, MROWS, BUFD, DD * TT, 1, stream);

    // ---- G2: [ai|aj|al] = cells @ Wpack1 (fp32) ----
    launch_gemm(false, cells_bf, UU, Wpk1_sw, AAL, 1536,
                nullptr, nullptr, 0, MROWS, 1536, UU, 0, stream);

    // ---- G3: bl = z_b @ W_l1[U:] (fp32); A = xh cols 0..1679 ----
    launch_gemm(false, xh_bf, GIN, Wl1b_sw, bl_f, HH,
                nullptr, nullptr, 0, MROWS, HH, BUFD, 0, stream);

    // ---- pairwise layer-1 ----
    {
        long long tot = (long long)PROWS * HH;
        pairwise_k<<<(int)((tot + 255) / 256), 256, 0, stream>>>(r1_bf, l1_bf, AAL, bl_f, b_r1, b_l1);
    }

    // ---- rho/lambda layers 2 & 3 (l3 adds r3 in-place into inter) ----
    launch_gemm(true,  r1_bf, HH, Wr2_sw, r2_bf, HH, b_r2, nullptr, 0, PROWS, HH, HH, 2, stream);
    launch_gemm(false, r2_bf, HH, Wr3_sw, inter_f, HH, b_r3, nullptr, 0, PROWS, HH, HH, 2, stream);
    launch_gemm(true,  l1_bf, HH, Wl2_sw, l2_bf, HH, b_l2, nullptr, 0, PROWS, HH, HH, 2, stream);
    launch_gemm(false, l2_bf, HH, Wl3_sw, inter_f, HH, b_l3, inter_f, HH, PROWS, HH, HH, 2, stream);

    // ---- sum_j / max_j -> xh[:,1680:2704] ----
    {
        long long tot = (long long)MROWS * HH;
        reduce_inter_k<<<(int)((tot + 255) / 256), 256, 0, stream>>>(xh_bf, inter_f);
    }

    // ---- gates: G = xh @ [W_I|W_F|W_O|W_C] ----
    launch_gemm(false, xh_bf, GIN, Wg_sw, G_f, NG,
                nullptr, nullptr, 0, MROWS, NG, GIN, 0, stream);

    // ---- LSTM pointwise ----
    {
        long long tot = (long long)MROWS * UU;
        lstm_point_k<<<(int)((tot + 255) / 256), 256, 0, stream>>>(
            out_C, out_H, Hn_bf, G_f, Cells, b_I, b_F, b_O, b_C);
    }

    // ---- decode ----
    launch_gemm(false, Hn_bf, UU, Wd_sw, out_dec, DD,
                b_d, nullptr, 0, MROWS, DD, UU, 0, stream);
}
